// Yolov5Loss_72464688218651
// MI455X (gfx1250) — compile-verified, matrix operand order checked by reading, not run on hardware
//
#include <hip/hip_runtime.h>
#include <cmath>

// ---------------------------------------------------------------------------
// YOLOv5 loss for MI455X (gfx1250, wave32).
// Passes: zero(ws) -> per-scale sample/scatter -> per-scale obj BCE reduce ->
//         per-scale cls BCE reduce -> finalize.
// Reductions accumulate through V_WMMA_F32_16X16X4_F32 with an all-ones B
// matrix: one WMMA folds 64 fp32 summands per wave into the fp32 C tile.
// Reduce loops are fully branchless (clamped b128 loads + one quad mask) so
// EXEC stays all-ones throughout (ISA §7.12 WMMA requirement) and no exec
// save/restore churn surrounds the transcendentals.
// ---------------------------------------------------------------------------

typedef __attribute__((ext_vector_type(2)))  float    v2f;
typedef __attribute__((ext_vector_type(4)))  float    v4f;
typedef __attribute__((ext_vector_type(8)))  float    v8f;
typedef __attribute__((ext_vector_type(16))) _Float16 v16h;

__device__ __forceinline__ float bce_logits(float x, float z) {
    // max(x,0) - x*z + log1p(exp(-|x|)); arg of log is in (1,2] -> plain log ok.
    return fmaxf(x, 0.0f) - x * z + __logf(1.0f + __expf(-fabsf(x)));
}

__device__ __forceinline__ float fast_sigmoid(float x) {
    return 1.0f / (1.0f + __expf(-x));
}

// Accumulate two fp32 summands per lane into a WMMA accumulator tile.
// B = ones -> D[m][n] = sum_k A[m][k] + C[m][n]; total tile sum == sum of all
// 64 per-wave inputs, regardless of which A slot each lane's values land in.
__device__ __forceinline__ void wmma_acc2(v8f& c, float e0, float e1) {
#if __has_builtin(__builtin_amdgcn_wmma_f32_16x16x4_f32)
    v2f a; a.x = e0;   a.y = e1;
    v2f b; b.x = 1.0f; b.y = 1.0f;
    c = __builtin_amdgcn_wmma_f32_16x16x4_f32(false, a, false, b,
                                              (short)0, c, false, false);
#else
    v16h a = {};
    a[0] = (_Float16)e0;
    a[1] = (_Float16)e1;
    v16h b;
#pragma unroll
    for (int i = 0; i < 16; ++i) b[i] = (_Float16)1.0f;
    c = __builtin_amdgcn_wmma_f32_16x16x32_f16(false, a, false, b,
                                               (short)0, c, false, false);
#endif
}

// Block-level finish: C/D layout => lanes 0-15 hold rows 0-7 (columns
// replicated), lanes 16-31 hold rows 8-15. Lane 0 + lane 16 of each wave
// carry the two half-sums of that wave's tile. `extra` folds in any scalar
// tail contribution (nonzero on at most one thread).
__device__ __forceinline__ void block_reduce_commit(v8f c, float extra,
                                                    float* accum) {
    __shared__ float red[16];                 // 8 waves * 2 halves
    float s = c[0] + c[1] + c[2] + c[3] + c[4] + c[5] + c[6] + c[7] + extra;
    int lane = threadIdx.x & 31;
    int wave = threadIdx.x >> 5;
    if (lane == 0)  red[wave * 2 + 0] = s;
    if (lane == 16) red[wave * 2 + 1] = s;
    __syncthreads();
    if (threadIdx.x == 0) {
        float t = 0.0f;
        int nw = (blockDim.x + 31) >> 5;
#pragma unroll
        for (int i = 0; i < 16; ++i)
            if (i < nw * 2) t += red[i];
        atomicAdd(accum, t);
    }
}

// ---------------------------------------------------------------------------
__global__ void zero_kernel(float* p, int n) {
    int stride = gridDim.x * blockDim.x;
    for (int i = blockIdx.x * blockDim.x + threadIdx.x; i < n; i += stride)
        p[i] = 0.0f;
}

// Per-sample gather: pred box from logits, IoU vs gt, loc accumulation,
// tobj scatter (plain store: duplicate-index behavior matches JAX .set race),
// and row offsets for the cls pass.
__global__ void sample_kernel(const float* __restrict__ box,
                              const int* __restrict__ samples,
                              const int* __restrict__ aidx,
                              const int* __restrict__ gy,
                              const int* __restrict__ gx,
                              const float* __restrict__ anchors,
                              const float* __restrict__ gt,
                              int N, int G,
                              float* __restrict__ tobj,
                              int* __restrict__ rowoff,
                              float* __restrict__ loc_accum) {
    int i = blockIdx.x * blockDim.x + threadIdx.x;
    if (i >= N) return;
    int b = samples[i], a = aidx[i], y = gy[i], x = gx[i];
    int row = ((b * 3 + a) * G + y) * G + x;
    rowoff[i] = row;

    const v4f f = *(const v4f*)(box + (long long)row * 4);
    float s0 = fast_sigmoid(f.x);
    float s1 = fast_sigmoid(f.y);
    float s2 = fast_sigmoid(f.z);
    float s3 = fast_sigmoid(f.w);
    float px = s0 * 2.0f - 0.5f;
    float py = s1 * 2.0f - 0.5f;
    float pw = (s2 * 2.0f) * (s2 * 2.0f) * anchors[i * 2 + 0];
    float ph = (s3 * 2.0f) * (s3 * 2.0f) * anchors[i * 2 + 1];

    float ax0 = px - pw * 0.5f, ay0 = py - ph * 0.5f;
    float ax1 = px + pw * 0.5f, ay1 = py + ph * 0.5f;
    const v4f g = *(const v4f*)(gt + (long long)i * 4);
    float bx0 = g.x - g.z * 0.5f, by0 = g.y - g.w * 0.5f;
    float bx1 = g.x + g.z * 0.5f, by1 = g.y + g.w * 0.5f;

    float iw = fmaxf(fminf(ax1, bx1) - fmaxf(ax0, bx0), 0.0f);
    float ih = fmaxf(fminf(ay1, by1) - fmaxf(ay0, by0), 0.0f);
    float inter  = iw * ih;
    float area_a = (ax1 - ax0) * (ay1 - ay0);
    float area_b = (bx1 - bx0) * (by1 - by0);
    float iou = inter / (area_a + area_b - inter + 1e-7f);

    atomicAdd(loc_accum, 1.0f - iou);
    tobj[row] = fmaxf(iou, 0.0f);
}

// Full-tensor obj BCE reduction. Quads are all-or-nothing valid (body and i0
// are multiples of 4), so one multiplicative mask per quad keeps the loop
// straight-line; clamped loads are always in-bounds. Thread 0 sweeps the
// (normally empty) scalar tail.
__global__ void obj_reduce_kernel(const float* __restrict__ obj,
                                  const float* __restrict__ tobj,
                                  int M, float* __restrict__ accum) {
    v8f c = {};
    const int body     = M & ~3;
    const int nthreads = gridDim.x * blockDim.x;
    const int per_iter = nthreads * 4;
    const int iters    = (body + per_iter - 1) / per_iter;
    const int gtid     = blockIdx.x * blockDim.x + threadIdx.x;
    const int last     = body >= 4 ? body - 4 : 0;
    for (int it = 0; it < iters; ++it) {
        int i0  = (it * nthreads + gtid) * 4;
        int i0c = i0 < last ? i0 : last;          // no-op for valid quads
        v4f x = *(const v4f*)(obj + i0c);
        v4f t = *(const v4f*)(tobj + i0c);
        float m   = (i0 < body) ? 1.0f : 0.0f;    // whole-quad validity
        float e01 = bce_logits(x.x, t.x) + bce_logits(x.y, t.y);
        float e23 = bce_logits(x.z, t.z) + bce_logits(x.w, t.w);
        wmma_acc2(c, m * e01, m * e23);
    }
    float extra = 0.0f;
    if (gtid == 0)
        for (int i = body; i < M; ++i) extra += bce_logits(obj[i], tobj[i]);
    block_reduce_commit(c, extra, accum);
}

// Gathered cls BCE vs one-hot, same pattern. C is a multiple of 4 (C=80), so
// a quad never crosses a gathered row and the b128 load stays 16B-aligned.
__global__ void cls_reduce_kernel(const float* __restrict__ cls,
                                  const int* __restrict__ rowoff,
                                  const int* __restrict__ labels,
                                  int N, int C, float* __restrict__ accum) {
    v8f c = {};
    const int total    = N * C;
    const int body     = total & ~3;
    const int nthreads = gridDim.x * blockDim.x;
    const int per_iter = nthreads * 4;
    const int iters    = (body + per_iter - 1) / per_iter;
    const int gtid     = blockIdx.x * blockDim.x + threadIdx.x;
    const int last     = body >= 4 ? body - 4 : 0;
    for (int it = 0; it < iters; ++it) {
        int i0  = (it * nthreads + gtid) * 4;
        int i0c = i0 < last ? i0 : last;
        int n   = i0c / C;
        int cc  = i0c - n * C;
        int lab = labels[n];
        v4f x = *(const v4f*)(cls + (long long)rowoff[n] * C + cc);
        float m   = (i0 < body) ? 1.0f : 0.0f;
        float e01 = bce_logits(x.x, (cc + 0 == lab) ? 1.0f : 0.0f)
                  + bce_logits(x.y, (cc + 1 == lab) ? 1.0f : 0.0f);
        float e23 = bce_logits(x.z, (cc + 2 == lab) ? 1.0f : 0.0f)
                  + bce_logits(x.w, (cc + 3 == lab) ? 1.0f : 0.0f);
        wmma_acc2(c, m * e01, m * e23);
    }
    float extra = 0.0f;
    if (gtid == 0)
        for (int u = body; u < total; ++u) {
            int n = u / C, cc = u - n * C;
            float x = cls[(long long)rowoff[n] * C + cc];
            extra += bce_logits(x, (cc == labels[n]) ? 1.0f : 0.0f);
        }
    block_reduce_commit(c, extra, accum);
}

__global__ void finalize_kernel(const float* __restrict__ acc,
                                const int* __restrict__ image_width,
                                float nll, float nml, float nhl,
                                float mll, float mml, float mhl,
                                float C, float* __restrict__ out) {
    if (blockIdx.x == 0 && threadIdx.x == 0) {
        float loc = 0.05f * (acc[0] / nll + acc[3] / nml + acc[6] / nhl);
        float w   = (float)image_width[0] / 640.0f;
        float obj = (w * w) * (4.0f * acc[1] / mll +
                               1.0f * acc[4] / mml +
                               0.4f * acc[7] / mhl);
        float cls = (0.5f * (C / 80.0f)) *
                    (acc[2] / (nll * C) + acc[5] / (nml * C) + acc[8] / (nhl * C));
        out[0] = loc;
        out[1] = obj;
        out[2] = cls;
    }
}

// ---------------------------------------------------------------------------
extern "C" void kernel_launch(void* const* d_in, const int* in_sizes, int n_in,
                              void* d_out, int out_size, void* d_ws, size_t ws_size,
                              hipStream_t stream) {
    (void)n_in; (void)out_size; (void)ws_size;

    int Ns[3], Ms[3], Gs[3], C = 80;
    for (int s = 0; s < 3; ++s) {
        int base = s * 10;
        Ms[s] = in_sizes[base + 1];                 // obj elements = 16*3*G*G
        Ns[s] = in_sizes[base + 3];                 // gathered sample count
        C     = in_sizes[base + 2] / Ms[s];         // class count
        Gs[s] = (int)(sqrt((double)(Ms[s] / 48)) + 0.5);
    }
    const int Ntot = Ns[0] + Ns[1] + Ns[2];
    const int Mtot = Ms[0] + Ms[1] + Ms[2];

    // Workspace: [0..15] accumulators (acc[s*3+{0:loc,1:obj,2:cls}]),
    // then rowoff (Ntot ints), then tobj (Mtot floats).
    float* acc    = (float*)d_ws;
    int*   rowoff = (int*)(acc + 16);
    float* tobj   = acc + 16 + Ntot;
    const int zcount = 16 + Ntot + Mtot;            // zeroing rowoff is harmless

    zero_kernel<<<256, 256, 0, stream>>>(acc, zcount);

    int roff = 0, toff = 0;
    for (int s = 0; s < 3; ++s) {
        int base = s * 10;
        sample_kernel<<<(Ns[s] + 255) / 256, 256, 0, stream>>>(
            (const float*)d_in[base + 0],
            (const int*)d_in[base + 3], (const int*)d_in[base + 4],
            (const int*)d_in[base + 5], (const int*)d_in[base + 6],
            (const float*)d_in[base + 7], (const float*)d_in[base + 8],
            Ns[s], Gs[s], tobj + toff, rowoff + roff, acc + s * 3 + 0);
        roff += Ns[s];
        toff += Ms[s];
    }

    toff = 0;
    for (int s = 0; s < 3; ++s) {
        int base   = s * 10;
        int quads  = (Ms[s] + 3) / 4;
        int blocks = (quads + 255) / 256;
        if (blocks > 1024) blocks = 1024;
        if (blocks < 1)    blocks = 1;
        obj_reduce_kernel<<<blocks, 256, 0, stream>>>(
            (const float*)d_in[base + 1], tobj + toff, Ms[s], acc + s * 3 + 1);
        toff += Ms[s];
    }

    roff = 0;
    for (int s = 0; s < 3; ++s) {
        int base   = s * 10;
        int total  = Ns[s] * C;
        int quads  = (total + 3) / 4;
        int blocks = (quads + 255) / 256;
        if (blocks > 1024) blocks = 1024;
        if (blocks < 1)    blocks = 1;
        cls_reduce_kernel<<<blocks, 256, 0, stream>>>(
            (const float*)d_in[base + 2], rowoff + roff,
            (const int*)d_in[base + 9], Ns[s], C, acc + s * 3 + 2);
        roff += Ns[s];
    }

    finalize_kernel<<<1, 32, 0, stream>>>(
        acc, (const int*)d_in[30],
        (float)Ns[0], (float)Ns[1], (float)Ns[2],
        (float)Ms[0], (float)Ms[1], (float)Ms[2],
        (float)C, (float*)d_out);
}
